// PerformerWrapper_6201932776074
// MI455X (gfx1250) — matrix operand order repaired and checked by reading, not compiled, not run on hardware
//
#include <hip/hip_runtime.h>
#include <cstdint>

// ---------------------------------------------------------------------------
// Performer graph-transformer block for MI455X (gfx1250, wave32, WMMA).
// All dense GEMMs run through v_wmma_f32_16x16x32_bf16 (fp32 accumulate),
// with the B operand staged through LDS (double-buffered) via the CDNA5
// async global->LDS path (ASYNCcnt-tracked) and shared by the 8 waves of
// each block.
// ---------------------------------------------------------------------------

#define DEV __device__ __forceinline__

typedef __attribute__((ext_vector_type(16))) __bf16   v16bf;
typedef __attribute__((ext_vector_type(8)))  float    v8f;
typedef __attribute__((ext_vector_type(4)))  unsigned uv4;
typedef __attribute__((ext_vector_type(4)))  float    fv4;
typedef __attribute__((ext_vector_type(4)))  int      i4v;

union Frag  { uv4 u[2]; v16bf v; };
union Pack8 { unsigned short s[8]; uv4 u; };

static constexpr int Dm   = 512;
static constexpr int Bb   = 32;
static constexpr int MAXN = 2048;
static constexpr int Hh   = 8;
static constexpr int DH   = 64;
static constexpr int Mf   = 256;

#if defined(__has_builtin)
#if __has_builtin(__builtin_amdgcn_global_load_async_to_lds_b128) && \
    __has_builtin(__builtin_amdgcn_s_wait_asynccnt)
#define USE_ASYNC_LDS 1
#endif
#endif

DEV unsigned short f2bf(float x) {
  union { float f; unsigned u; } v; v.f = x;
  unsigned r = v.u + 0x7fffu + ((v.u >> 16) & 1u);
  return (unsigned short)(r >> 16);
}
DEV float bf2f(unsigned short h) {
  union { unsigned u; float f; } v; v.u = ((unsigned)h) << 16; return v.f;
}
DEV void atomicMaxF(float* addr, float val) {
  unsigned* ua = (unsigned*)addr;
  unsigned old = *ua;
  while (true) {
    union { unsigned u; float f; } c; c.u = old;
    if (c.f >= val) break;
    unsigned assumed = old;
    union { float f; unsigned u; } nv; nv.f = val;
    old = atomicCAS(ua, assumed, nv.u);
    if (old == assumed) break;
  }
}
DEV v8f zero8() { v8f z; for (int i = 0; i < 8; ++i) z[i] = 0.f; return z; }

DEV void stage_b128(const unsigned short* g, unsigned short* l) {
#if USE_ASYNC_LDS
  __builtin_amdgcn_global_load_async_to_lds_b128(
      (__attribute__((address_space(1))) i4v*)g,
      (__attribute__((address_space(3))) i4v*)l, 0, 0);
#else
  *(uv4*)l = *(const uv4*)g;
#endif
}
DEV void stage_wait() {
#if USE_ASYNC_LDS
  __builtin_amdgcn_s_wait_asynccnt(0);
#endif
}

// LDS B tile: 64 columns x 32 k of bf16, row pitch padded to 40 ushorts so
// the 16-byte fragment reads of 16 lanes hit disjoint bank quadruples.
static constexpr int BP        = 40;
static constexpr int BUFSTRIDE = 64 * BP;

// Modes:
// 0 conv: out = relu(acc + bias[col]) -> fout [rows x 512] f32
// 1 qk  : scale by 64^-0.25, store bf16 [b,h,n,dh]
// 3 v   : store bf16 transposed [b,h,dh,n]
// 4 phi pass1 query: atomicMax row maxes -> fout = rowmax[bh*2048+n]
// 5 phi pass1 key  : atomicMax global    -> fout = bhmax[bh]
// 6 phi pass2 query: qf = exp(u-diag-rowmax)/16 + 1e-6 -> uout bf16 [bh,n,m]
// 7 phi pass2 key  : masked, store kft bf16 [bh,m,n], atomicAdd ksum[bh,m]
// 8 kv  : store bf16 kvT [bh,dh,m]
// 9 attn: val*z[bh,n] -> attn_dense bf16 [b,n, h*64+dh]
// 10 wo : store f32 fout [rows x 512]
template <int MODE>
__global__ __launch_bounds__(256) void gemm_k(
    const unsigned short* __restrict__ A, int lda,
    const unsigned short* __restrict__ BT, int ldbt,
    int K, int rowsTotal,
    const float* __restrict__ f0, const float* __restrict__ f1,
    const int* __restrict__ cnts,
    float* __restrict__ fout, unsigned short* __restrict__ uout,
    float* __restrict__ ksum) {
  __shared__ unsigned short ldsB[2 * BUFSTRIDE];

  int tid  = threadIdx.x;
  int lane = tid & 31;
  int wave = tid >> 5;          // 0..7
  int bh   = blockIdx.z;

  if constexpr (MODE >= 4 && MODE <= 7) { A += (long)bh * MAXN * DH; }
  if constexpr (MODE == 8) { A += (long)bh * Mf * MAXN; BT += (long)bh * DH * MAXN; }
  if constexpr (MODE == 9) { A += (long)bh * MAXN * Mf; BT += (long)bh * DH * Mf; }

  int rowBase = blockIdx.y * 128 + wave * 16;
  int colBase = blockIdx.x * 64;
  int maxRow  = rowsTotal - 1;

  // A stream (per-wave, direct from global; CDNA5 ISA 7.12.2 A layout)
  int r = rowBase + (lane & 15);
  if (r > maxRow) r = maxRow;
  const unsigned short* ap = A + (long)r * lda + ((lane < 16) ? 0 : 8);

  // B tile staging indices: 256 threads x 16B = 4 KB tile per k-step
  int lcol  = tid >> 2;         // 0..63
  int lpart = tid & 3;          // 0..3
  const unsigned short* bsrc = BT + (long)(colBase + lcol) * ldbt + lpart * 8;
  unsigned short*       bdst = &ldsB[lcol * BP + lpart * 8];

  int kb = (lane < 16) ? 0 : 8;
  int cl = lane & 15;
  int nk = K >> 5;

  v8f acc[4]; for (int j = 0; j < 4; ++j) acc[j] = zero8();

  stage_b128(bsrc, bdst);       // prologue: k-step 0 into buffer 0
  stage_wait();
  __syncthreads();

  for (int s = 0; s < nk; ++s) {
    int cur = s & 1;
    if (s + 1 < nk)
      stage_b128(bsrc + (s + 1) * 32, bdst + ((s + 1) & 1) * BUFSTRIDE);

    __builtin_prefetch(ap + 256, 0, 1);
    Frag fa;
    fa.u[0] = *(const uv4*)(ap);
    fa.u[1] = *(const uv4*)(ap + 16);
    ap += 32;

    const unsigned short* lb = &ldsB[cur * BUFSTRIDE];
#pragma unroll
    for (int j = 0; j < 4; ++j) {
      const unsigned short* bp = lb + (j * 16 + cl) * BP + kb;
      Frag fb;
      fb.u[0] = *(const uv4*)bp;
      fb.u[1] = *(const uv4*)(bp + 16);
      acc[j] = __builtin_amdgcn_wmma_f32_16x16x32_bf16(
          false, fa.v, false, fb.v, (short)0, acc[j], false, false);
    }
    stage_wait();
    __syncthreads();
  }

  int roff = (lane < 16) ? 0 : 8;

  if constexpr (MODE == 0 || MODE == 10) {
    for (int j = 0; j < 4; ++j)
      for (int vr = 0; vr < 8; ++vr) {
        int row = rowBase + roff + vr;
        int col = colBase + j * 16 + cl;
        if (row < rowsTotal) {
          float v = acc[j][vr];
          if constexpr (MODE == 0) v = fmaxf(v + f0[col], 0.f);
          fout[(long)row * Dm + col] = v;
        }
      }
  } else if constexpr (MODE == 1 || MODE == 3) {
    for (int j = 0; j < 4; ++j)
      for (int vr = 0; vr < 8; ++vr) {
        int row = rowBase + roff + vr;
        int col = colBase + j * 16 + cl;
        if (row >= rowsTotal) continue;
        int b = row >> 11, n = row & (MAXN - 1);
        int h = col >> 6,  dh = col & (DH - 1);
        float v = acc[j][vr];
        if constexpr (MODE == 1) {
          v *= 0.35355339059327373f;  // 64^-0.25
          uout[(((long)(b * Hh + h) * MAXN) + n) * DH + dh] = f2bf(v);
        } else {
          uout[(((long)(b * Hh + h) * DH) + dh) * MAXN + n] = f2bf(v);
        }
      }
  } else if constexpr (MODE == 4) {
    float mrow[8];
    for (int vr = 0; vr < 8; ++vr) {
      float m = acc[0][vr];
      m = fmaxf(m, acc[1][vr]); m = fmaxf(m, acc[2][vr]); m = fmaxf(m, acc[3][vr]);
      for (int s = 1; s < 16; s <<= 1) m = fmaxf(m, __shfl_xor(m, s, 32));
      mrow[vr] = m;
    }
    if (cl == 0)
      for (int vr = 0; vr < 8; ++vr)
        atomicMaxF(&fout[(long)bh * MAXN + rowBase + roff + vr], mrow[vr]);
  } else if constexpr (MODE == 5) {
    float m = -3.0e38f;
    for (int j = 0; j < 4; ++j)
      for (int vr = 0; vr < 8; ++vr) m = fmaxf(m, acc[j][vr]);
    for (int s = 1; s < 32; s <<= 1) m = fmaxf(m, __shfl_xor(m, s, 32));
    if (lane == 0) atomicMaxF(&fout[bh], m);
  } else if constexpr (MODE == 6) {
    float dgv[8], mxv[8];
    for (int vr = 0; vr < 8; ++vr) {
      int row = rowBase + roff + vr;
      dgv[vr] = f0[(long)bh * MAXN + row];
      mxv[vr] = f1[(long)bh * MAXN + row];
    }
    for (int j = 0; j < 4; ++j)
      for (int vr = 0; vr < 8; ++vr) {
        int row = rowBase + roff + vr;
        int col = colBase + j * 16 + cl;
        float e = __expf(acc[j][vr] - dgv[vr] - mxv[vr]) * 0.0625f + 1e-6f;
        uout[((long)bh * MAXN + row) * Mf + col] = f2bf(e);
      }
  } else if constexpr (MODE == 7) {
    int b = bh >> 3;
    int cnt = cnts[b];
    float mx = f1[bh];
    float dgv[8];
    for (int vr = 0; vr < 8; ++vr)
      dgv[vr] = f0[(long)bh * MAXN + rowBase + roff + vr];
    for (int j = 0; j < 4; ++j) {
      int col = colBase + j * 16 + cl;
      Pack8 pk; float psum = 0.f;
      for (int vr = 0; vr < 8; ++vr) {
        int row = rowBase + roff + vr;
        float e = __expf(acc[j][vr] - dgv[vr] - mx) * 0.0625f + 1e-6f;
        if (row >= cnt) e = 0.f;
        pk.s[vr] = f2bf(e); psum += e;
      }
      *(uv4*)(uout + ((long)bh * Mf + col) * MAXN + rowBase + roff) = pk.u;
      psum += __shfl_xor(psum, 16, 32);
      if (lane < 16) atomicAdd(&ksum[(long)bh * Mf + colBase + j * 16 + lane], psum);
    }
  } else if constexpr (MODE == 8) {
    for (int j = 0; j < 4; ++j)
      for (int vr = 0; vr < 8; ++vr) {
        int row = rowBase + roff + vr;        // m
        int col = colBase + j * 16 + cl;      // dh
        uout[((long)bh * DH + col) * Mf + row] = f2bf(acc[j][vr]);
      }
  } else if constexpr (MODE == 9) {
    int b = bh >> 3, h = bh & 7;
    for (int j = 0; j < 4; ++j)
      for (int vr = 0; vr < 8; ++vr) {
        int row = rowBase + roff + vr;        // n
        int col = colBase + j * 16 + cl;      // dh
        float z = f0[(long)bh * MAXN + row];
        uout[((long)b * MAXN + row) * Dm + h * DH + col] = f2bf(acc[j][vr] * z);
      }
  }
}

// ---------------------------------------------------------------------------
// Elementwise / reduction helpers
// ---------------------------------------------------------------------------
__global__ void fill_f32_k(float* p, float v, long n) {
  long i = (long)blockIdx.x * blockDim.x + threadIdx.x;
  if (i < n) p[i] = v;
}
__global__ void bn_sum_k(const float* __restrict__ X, int rows,
                         float* __restrict__ sums) {
  int t = threadIdx.x;                 // 256
  long r0 = (long)blockIdx.x * 256;
  long r1 = r0 + 256; if (r1 > rows) r1 = rows;
  float s0 = 0, s1 = 0, q0 = 0, q1 = 0;
  for (long r = r0; r < r1; ++r) {
    float a = X[r * Dm + t];
    float b = X[r * Dm + t + 256];
    s0 += a; q0 += a * a; s1 += b; q1 += b * b;
  }
  atomicAdd(&sums[t], s0);        atomicAdd(&sums[t + 256], s1);
  atomicAdd(&sums[512 + t], q0);  atomicAdd(&sums[512 + t + 256], q1);
}
__global__ void bn_fin_k(const float* sums, float invN, float* stats) {
  int d = threadIdx.x;                 // 512
  float mu = sums[d] * invN;
  float var = sums[512 + d] * invN - mu * mu;
  stats[d] = mu;
  stats[512 + d] = rsqrtf(var + 1e-5f);
}
__global__ void xn_k(const float* x, const float* stats, const float* g,
                     const float* b, float* xn, long n) {
  long i = (long)blockIdx.x * blockDim.x + threadIdx.x;
  if (i >= n) return;
  int d = (int)(i & (Dm - 1));
  xn[i] = (x[i] - stats[d]) * stats[512 + d] * g[d] + b[d];
}
__global__ void trans_k(const float* W, unsigned short* WT) {
  long i = (long)blockIdx.x * blockDim.x + threadIdx.x;
  if (i >= (long)Dm * Dm) return;
  int k = (int)(i & (Dm - 1)), nn = (int)(i >> 9);
  WT[i] = f2bf(W[(long)k * Dm + nn]);
}
__global__ void cvt_k(const float* s, unsigned short* d, long n) {
  long i = (long)blockIdx.x * blockDim.x + threadIdx.x;
  if (i < n) d[i] = f2bf(s[i]);
}
__global__ void deg_k(const int* ei, int E, float* deg) {
  long e = (long)blockIdx.x * blockDim.x + threadIdx.x;
  if (e < E) atomicAdd(&deg[ei[E + e]], 1.0f);
}
__global__ void scat_k(const int* ei, int E, const float* xn, float* agg) {
  long t = (long)blockIdx.x * blockDim.x + threadIdx.x;
  if (t >= (long)E * 128) return;
  long e = t >> 7;
  int c = (int)(t & 127) * 4;
  int src = ei[e], dst = ei[E + e];
  fv4 v = *(const fv4*)(xn + (long)src * Dm + c);
  float* a = agg + (long)dst * Dm + c;
  atomicAdd(a + 0, v[0]); atomicAdd(a + 1, v[1]);
  atomicAdd(a + 2, v[2]); atomicAdd(a + 3, v[3]);
}
__global__ void cnt_k(const int* batch, int N, int* counts) {
  long i = (long)blockIdx.x * blockDim.x + threadIdx.x;
  if (i < N) atomicAdd(&counts[batch[i]], 1);
}
__global__ void off_k(const int* counts, int* offs) {
  if (threadIdx.x == 0 && blockIdx.x == 0) {
    int acc = 0;
    for (int b = 0; b < Bb; ++b) { offs[b] = acc; acc += counts[b]; }
  }
}
__global__ void pos_k(const int* batch, const int* offs, int N, int* pos) {
  long i = (long)blockIdx.x * blockDim.x + threadIdx.x;
  if (i < N) pos[i] = (int)i - offs[batch[i]];
}
__global__ void hd_k(const float* x, const int* batch, const int* pos,
                     unsigned short* hd, long n) {
  long i = (long)blockIdx.x * blockDim.x + threadIdx.x;
  if (i >= n) return;
  long node = i >> 9; int d = (int)(i & (Dm - 1));
  int b = batch[node];
  hd[((long)b * MAXN + pos[node]) * Dm + d] = f2bf(x[i]);
}
__global__ void aggn_k(const float* agg, const float* deg, unsigned short* o,
                       long n) {
  long i = (long)blockIdx.x * blockDim.x + threadIdx.x;
  if (i >= n) return;
  long node = i >> 9;
  o[i] = f2bf(agg[i] / fmaxf(deg[node], 1.0f));
}
__global__ void diag_k(const unsigned short* T, float* diag, int n) {
  long t = (long)blockIdx.x * blockDim.x + threadIdx.x;
  if (t >= n) return;
  const unsigned short* p = T + (long)t * DH;
  float s = 0.f;
  for (int d = 0; d < DH; ++d) { float v = bf2f(p[d]); s += v * v; }
  diag[t] = 0.5f * s;
}
__global__ void zk_k(const unsigned short* qf, const float* ksum, float* z,
                     int n) {
  long t = (long)blockIdx.x * blockDim.x + threadIdx.x;
  if (t >= n) return;
  int bh = (int)(t >> 11);
  const unsigned short* q = qf + (long)t * Mf;
  const float* ks = ksum + (long)bh * Mf;
  float s = 0.f;
  for (int m = 0; m < Mf; ++m) s += bf2f(q[m]) * ks[m];
  z[t] = 1.0f / (s + 1e-6f);
}
__global__ void ypre_k(const float* conv, const float* attn, const float* x,
                       const int* batch, const int* pos, float* ypre, long n) {
  long i = (long)blockIdx.x * blockDim.x + threadIdx.x;
  if (i >= n) return;
  long node = i >> 9; int d = (int)(i & (Dm - 1));
  int b = batch[node];
  float at = attn[((long)b * MAXN + pos[node]) * Dm + d];
  ypre[i] = conv[i] + at + x[i];
}
__global__ void out_k(const float* ypre, const float* stats, const float* g,
                      const float* b, float* out, long n) {
  long i = (long)blockIdx.x * blockDim.x + threadIdx.x;
  if (i >= n) return;
  int d = (int)(i & (Dm - 1));
  out[i] = (ypre[i] - stats[d]) * stats[512 + d] * g[d] + b[d];
}

// ---------------------------------------------------------------------------
// Host launcher.  Requires ws_size >= ~1.5 GB for N = 48544.
// ---------------------------------------------------------------------------
extern "C" void kernel_launch(void* const* d_in, const int* in_sizes, int n_in,
                              void* d_out, int out_size, void* d_ws,
                              size_t ws_size, hipStream_t stream) {
  const float* x     = (const float*)d_in[0];
  const int*   batch = (const int*)  d_in[1];
  const int*   ei    = (const int*)  d_in[2];
  const float* g1 = (const float*)d_in[3];
  const float* b1 = (const float*)d_in[4];
  const float* g2 = (const float*)d_in[5];
  const float* b2 = (const float*)d_in[6];
  const float* Wc = (const float*)d_in[7];
  const float* bc = (const float*)d_in[8];
  const float* Wq = (const float*)d_in[9];
  const float* Wk = (const float*)d_in[10];
  const float* Wv = (const float*)d_in[11];
  const float* Wo = (const float*)d_in[12];
  const float* pr = (const float*)d_in[13];

  const int N = in_sizes[0] / Dm;
  const int E = in_sizes[2] / 2;
  const long fN = (long)N * Dm;
  const long DENSE = (long)Bb * MAXN * Dm;          // 33,554,432
  const long BHN   = (long)Bb * Hh * MAXN;          // 524,288

  char* base = (char*)d_ws;
  size_t cur = 0;
  auto carve = [&](size_t bytes) -> void* {
    void* p = base + cur;
    cur += (bytes + 255) & ~(size_t)255;
    return p;
  };
  float* sums   = (float*)carve(1024 * 4);
  float* stats1 = (float*)carve(1024 * 4);
  float* stats2 = (float*)carve(1024 * 4);
  float* xn     = (float*)carve(fN * 4);
  float* agg    = (float*)carve(fN * 4);
  float* deg    = (float*)carve((size_t)N * 4);
  int*   counts = (int*)  carve(Bb * 4);
  int*   offs   = (int*)  carve(Bb * 4);
  int*   pos    = (int*)  carve((size_t)N * 4);
  unsigned short* WcT = (unsigned short*)carve((size_t)Dm * Dm * 2);
  unsigned short* WqT = (unsigned short*)carve((size_t)Dm * Dm * 2);
  unsigned short* WkT = (unsigned short*)carve((size_t)Dm * Dm * 2);
  unsigned short* WvT = (unsigned short*)carve((size_t)Dm * Dm * 2);
  unsigned short* WoT = (unsigned short*)carve((size_t)Dm * Dm * 2);
  unsigned short* prb = (unsigned short*)carve((size_t)Mf * DH * 2);
  unsigned short* aggn = (unsigned short*)carve(fN * 2);
  float* conv   = (float*)carve(fN * 4);
  unsigned short* hd  = (unsigned short*)carve(DENSE * 2);
  unsigned short* qb  = (unsigned short*)carve(DENSE * 2);
  unsigned short* kb  = (unsigned short*)carve(DENSE * 2);
  unsigned short* vT  = (unsigned short*)carve(DENSE * 2);
  float* diag_q = (float*)carve(BHN * 4);
  float* diag_kv= (float*)carve(BHN * 4);
  float* rowmax = (float*)carve(BHN * 4);
  float* bhmax  = (float*)carve((size_t)Bb * Hh * 4);
  unsigned short* qf  = (unsigned short*)carve((size_t)BHN * Mf * 2);   // 268 MB
  unsigned short* kft = (unsigned short*)carve((size_t)BHN * Mf * 2);   // 268 MB
  float* ksum   = (float*)carve((size_t)Bb * Hh * Mf * 4);
  unsigned short* kvT = (unsigned short*)carve((size_t)Bb * Hh * DH * Mf * 2);
  float* zbuf   = (float*)carve(BHN * 4);
  unsigned short* attn_d = (unsigned short*)carve(DENSE * 2);
  float* attn_o = (float*)carve(DENSE * 4);
  float* ypre   = (float*)carve(fN * 4);
  (void)ws_size; (void)n_in; (void)out_size;

  auto gridFor = [](long n, int blk) { return dim3((unsigned)((n + blk - 1) / blk)); };

  // ---- zero init ----
  (void)hipMemsetAsync(sums, 0, 1024 * 4, stream);
  (void)hipMemsetAsync(agg, 0, fN * 4, stream);
  (void)hipMemsetAsync(deg, 0, (size_t)N * 4, stream);
  (void)hipMemsetAsync(counts, 0, Bb * 4, stream);
  (void)hipMemsetAsync(hd, 0, DENSE * 2, stream);
  (void)hipMemsetAsync(ksum, 0, (size_t)Bb * Hh * Mf * 4, stream);

  // ---- batchnorm-1 stats + apply ----
  bn_sum_k<<<gridFor(N, 256), 256, 0, stream>>>(x, N, sums);
  bn_fin_k<<<1, 512, 0, stream>>>(sums, 1.0f / (float)N, stats1);
  xn_k<<<gridFor(fN, 256), 256, 0, stream>>>(x, stats1, g1, b1, xn, fN);

  // ---- weight transpose + convert to bf16 ----
  trans_k<<<gridFor((long)Dm * Dm, 256), 256, 0, stream>>>(Wc, WcT);
  trans_k<<<gridFor((long)Dm * Dm, 256), 256, 0, stream>>>(Wq, WqT);
  trans_k<<<gridFor((long)Dm * Dm, 256), 256, 0, stream>>>(Wk, WkT);
  trans_k<<<gridFor((long)Dm * Dm, 256), 256, 0, stream>>>(Wv, WvT);
  trans_k<<<gridFor((long)Dm * Dm, 256), 256, 0, stream>>>(Wo, WoT);
  cvt_k<<<gridFor((long)Mf * DH, 256), 256, 0, stream>>>(pr, prb, (long)Mf * DH);

  // ---- GCN aggregation ----
  deg_k<<<gridFor(E, 256), 256, 0, stream>>>(ei, E, deg);
  scat_k<<<gridFor((long)E * 128, 256), 256, 0, stream>>>(ei, E, xn, agg);
  aggn_k<<<gridFor(fN, 256), 256, 0, stream>>>(agg, deg, aggn, fN);

  // ---- dense batch ----
  cnt_k<<<gridFor(N, 256), 256, 0, stream>>>(batch, N, counts);
  off_k<<<1, 32, 0, stream>>>(counts, offs);
  pos_k<<<gridFor(N, 256), 256, 0, stream>>>(batch, offs, N, pos);
  hd_k<<<gridFor(fN, 256), 256, 0, stream>>>(x, batch, pos, hd, fN);

  // ---- conv GEMM: relu(aggn @ Wc + bc) ----
  gemm_k<0><<<dim3(8, (N + 127) / 128, 1), 256, 0, stream>>>(
      aggn, Dm, WcT, Dm, Dm, N, bc, nullptr, nullptr, conv, nullptr, nullptr);

  // ---- Q/K/V GEMMs ----
  const int denseRows = Bb * MAXN;
  gemm_k<1><<<dim3(8, denseRows / 128, 1), 256, 0, stream>>>(
      hd, Dm, WqT, Dm, Dm, denseRows, nullptr, nullptr, nullptr, nullptr, qb, nullptr);
  gemm_k<1><<<dim3(8, denseRows / 128, 1), 256, 0, stream>>>(
      hd, Dm, WkT, Dm, Dm, denseRows, nullptr, nullptr, nullptr, nullptr, kb, nullptr);
  gemm_k<3><<<dim3(8, denseRows / 128, 1), 256, 0, stream>>>(
      hd, Dm, WvT, Dm, Dm, denseRows, nullptr, nullptr, nullptr, nullptr, vT, nullptr);

  // ---- diag terms ----
  diag_k<<<gridFor(BHN, 256), 256, 0, stream>>>(qb, diag_q, (int)BHN);
  diag_k<<<gridFor(BHN, 256), 256, 0, stream>>>(kb, diag_kv, (int)BHN);

  // ---- phi pass 1: maxes ----
  fill_f32_k<<<gridFor(BHN, 256), 256, 0, stream>>>(rowmax, -3.0e38f, BHN);
  fill_f32_k<<<1, 256, 0, stream>>>(bhmax, -3.0e38f, Bb * Hh);
  gemm_k<4><<<dim3(Mf / 64, MAXN / 128, Bb * Hh), 256, 0, stream>>>(
      qb, DH, prb, DH, DH, MAXN, nullptr, nullptr, nullptr, rowmax, nullptr, nullptr);
  gemm_k<5><<<dim3(Mf / 64, MAXN / 128, Bb * Hh), 256, 0, stream>>>(
      kb, DH, prb, DH, DH, MAXN, nullptr, nullptr, nullptr, bhmax, nullptr, nullptr);

  // ---- phi pass 2: exponentials ----
  gemm_k<6><<<dim3(Mf / 64, MAXN / 128, Bb * Hh), 256, 0, stream>>>(
      qb, DH, prb, DH, DH, MAXN, diag_q, rowmax, nullptr, nullptr, qf, nullptr);
  gemm_k<7><<<dim3(Mf / 64, MAXN / 128, Bb * Hh), 256, 0, stream>>>(
      kb, DH, prb, DH, DH, MAXN, diag_kv, bhmax, counts, nullptr, kft, ksum);

  // ---- z = 1/(qf . ksum + eps) ----
  zk_k<<<gridFor(BHN, 256), 256, 0, stream>>>(qf, ksum, zbuf, (int)BHN);

  // ---- kv = kft @ vm^T  (per b,h) ----
  gemm_k<8><<<dim3(1, Mf / 128, Bb * Hh), 256, 0, stream>>>(
      kft, MAXN, vT, MAXN, MAXN, Mf, nullptr, nullptr, nullptr, nullptr, kvT, nullptr);

  // ---- attn = (qf @ kv) * z ----
  gemm_k<9><<<dim3(1, MAXN / 128, Bb * Hh), 256, 0, stream>>>(
      qf, Mf, kvT, Mf, Mf, MAXN, zbuf, nullptr, nullptr, nullptr, attn_d, nullptr);

  // ---- output projection Wo ----
  gemm_k<10><<<dim3(8, denseRows / 128, 1), 256, 0, stream>>>(
      attn_d, Dm, WoT, Dm, Dm, denseRows, nullptr, nullptr, nullptr, attn_o,
      nullptr, nullptr);

  // ---- residual + batchnorm-2 ----
  ypre_k<<<gridFor(fN, 256), 256, 0, stream>>>(conv, attn_o, x, batch, pos, ypre, fN);
  (void)hipMemsetAsync(sums, 0, 1024 * 4, stream);
  bn_sum_k<<<gridFor(N, 256), 256, 0, stream>>>(ypre, N, sums);
  bn_fin_k<<<1, 512, 0, stream>>>(sums, 1.0f / (float)N, stats2);
  out_k<<<gridFor(fN, 256), 256, 0, stream>>>(ypre, stats2, g2, b2, (float*)d_out, fN);
}
